// NetVLAD_38886633898451
// MI455X (gfx1250) — matrix operand order repaired and checked by reading, not compiled, not run on hardware
//
#include <hip/hip_runtime.h>

// ---------------------------------------------------------------------------
// NetVLAD head for MI455X (gfx1250).
// Convs = implicit GEMMs on the bf16 WMMA pipe with a 3-term bf16 split
// (hi*hi + hi*lo + lo*hi, fp32 accum) for near-fp32 accuracy at bf16 matrix
// rates. Hi/lo split is precomputed (or fused into the producer epilogue) so
// the GEMM hot loop moves only 16-bit data. LDS tiles are laid out so every
// WMMA fragment is two aligned ds_load_b128 (80B row stride, conflict-free
// banks); LDS is double-buffered with ONE barrier per K-step; the weight tile
// is staged with GLOBAL_LOAD_ASYNC_TO_LDS (ASYNCcnt) when available.
// ---------------------------------------------------------------------------

typedef __attribute__((ext_vector_type(16))) __bf16 v16bf;
typedef __attribute__((ext_vector_type(8)))  float  v8f;

#define CIN   512
#define COUT  512
#define NPIX  16384      // 16 * 32 * 32
#define KRED  4608       // 512 * 9
#define KSTEP 32
#define MT    128        // workgroup M tile (Cout)
#define NT    128        // workgroup N tile (pixels)
#define LDSK  40         // padded K stride in ushorts (80B rows, bank-friendly)
#define L2EPS 1e-12f

#if defined(__has_builtin)
#  if __has_builtin(__builtin_amdgcn_global_load_async_to_lds_b32)
#    define HAVE_ASYNC_LDS 1
#  endif
#  if __has_builtin(__builtin_amdgcn_s_wait_asynccnt)
#    define HAVE_WAIT_ASYNC 1
#  endif
#endif

#if defined(HAVE_ASYNC_LDS)
// builtin signature (from hipcc diagnostics): (__device__ int*, __shared__ int*, Ii, Ii)
typedef __attribute__((address_space(1))) int gint_t;
typedef __attribute__((address_space(3))) int lint_t;
#endif

__device__ __forceinline__ void wait_async0() {
#if defined(HAVE_WAIT_ASYNC)
  __builtin_amdgcn_s_wait_asynccnt(0);
#elif defined(HAVE_ASYNC_LDS)
  asm volatile("s_wait_asynccnt 0x0" ::: "memory");
#endif
}

union FragU { v16bf v; uint4 q[2]; };

__device__ __forceinline__ unsigned short f2bf_rne(float f) {
  unsigned int u = __float_as_uint(f);
  unsigned int r = u + 0x7FFFu + ((u >> 16) & 1u);   // round-to-nearest-even
  return (unsigned short)(r >> 16);
}
__device__ __forceinline__ float bf2f(unsigned short h) {
  return __uint_as_float(((unsigned int)h) << 16);
}

// ---------------------------------------------------------------------------
// fp32 -> (bf16 hi, bf16 lo) split, elementwise prep (memory bound)
// ---------------------------------------------------------------------------
__global__ __launch_bounds__(256)
void split_bf16(const float* __restrict__ src, unsigned short* __restrict__ hi,
                unsigned short* __restrict__ lo, int nElem)
{
  int i = blockIdx.x * 256 + threadIdx.x;
  if (i < nElem) {
    float f = src[i];
    unsigned short h = f2bf_rne(f);
    hi[i] = h;
    lo[i] = f2bf_rne(f - bf2f(h));
  }
}

// ---------------------------------------------------------------------------
// conv3x3 (pad=1) as implicit GEMM on pre-split bf16 hi/lo operands:
//   out[m, p] = sum_r W[m, r] * im2col[r, p]
// WG = 256 threads = 8 waves; WG tile 128x128; wave tile 32x64 (2x4 frags).
// ---------------------------------------------------------------------------
__global__ __launch_bounds__(256)
void conv3x3_wmma(const unsigned short* __restrict__ inHi,
                  const unsigned short* __restrict__ inLo,
                  const unsigned short* __restrict__ wHi,
                  const unsigned short* __restrict__ wLo,
                  float* __restrict__ outF,
                  unsigned short* __restrict__ outHi,
                  unsigned short* __restrict__ outLo,
                  const float* __restrict__ g, const float* __restrict__ b,
                  const float* __restrict__ mu, const float* __restrict__ var,
                  int bn_relu_split)
{
  __shared__ alignas(16) unsigned short sA[2][2][MT][LDSK]; // [buf][hi/lo][m][k]
  __shared__ alignas(16) unsigned short sB[2][2][NT][LDSK]; // [buf][hi/lo][p][k]

  const int tid    = threadIdx.x;
  const int lane   = tid & 31;
  const int wave   = tid >> 5;
  const int mBase  = blockIdx.x * MT;
  const int nBase  = blockIdx.y * NT;
  const int mW     = (wave & 3) * 32;   // 4 waves along M
  const int nW     = (wave >> 2) * 64;  // 2 waves along N
  const int hiHalf = (lane >= 16) ? 1 : 0;
  const int rlane  = lane & 15;

  v8f acc[2][4] = {};

  unsigned short rBh[16], rBl[16];      // B prefetch regs (im2col gather)
#if !defined(HAVE_ASYNC_LDS)
  unsigned int rAh[8], rAl[8];          // A prefetch regs (fallback path)
#endif

#if defined(HAVE_ASYNC_LDS)
  // A tile: contiguous 16-bit copy -> async global->LDS (ASYNCcnt tracked)
  auto issueA = [&](int kb, int bufw) {
#pragma unroll
    for (int i = 0; i < 8; ++i) {
      int e  = i * 256 + tid;
      int rr = e >> 4, kk = (e & 15) << 1;
      int gi = (mBase + rr) * KRED + kb + kk;     // even -> 4B aligned
      __builtin_amdgcn_global_load_async_to_lds_b32(
          (gint_t*)&wHi[gi], (lint_t*)&sA[bufw][0][rr][kk], 0, 0);
      __builtin_amdgcn_global_load_async_to_lds_b32(
          (gint_t*)&wLo[gi], (lint_t*)&sA[bufw][1][rr][kk], 0, 0);
    }
  };
#else
  auto issueA = [&](int kb, int) {
#pragma unroll
    for (int i = 0; i < 8; ++i) {
      int e  = i * 256 + tid;
      int rr = e >> 4, kk = (e & 15) << 1;
      int gi = (mBase + rr) * KRED + kb + kk;
      rAh[i] = *(const unsigned int*)&wHi[gi];
      rAl[i] = *(const unsigned int*)&wLo[gi];
    }
  };
#endif
  auto storeA = [&](int bufw) {
#if !defined(HAVE_ASYNC_LDS)
#pragma unroll
    for (int i = 0; i < 8; ++i) {
      int e  = i * 256 + tid;
      int rr = e >> 4, kk = (e & 15) << 1;
      *(unsigned int*)&sA[bufw][0][rr][kk] = rAh[i];
      *(unsigned int*)&sA[bufw][1][rr][kk] = rAl[i];
    }
#else
    (void)bufw;
#endif
  };
  auto loadB = [&](int kb) {
#pragma unroll
    for (int i = 0; i < 16; ++i) {      // B: im2col gather, 128x32 ushorts
      int e  = i * 256 + tid;
      int rr = e >> 5, kk = e & 31;
      int p  = nBase + rr;
      int r  = kb + kk;
      int ci = r / 9; int rem = r - ci * 9;
      int kh = rem / 3, kw = rem - kh * 3;
      int n  = p >> 10, hw = p & 1023;
      int hh = (hw >> 5) + kh - 1;
      int ww = (hw & 31) + kw - 1;
      if ((unsigned)hh < 32u && (unsigned)ww < 32u) {
        int gi = ((n * CIN + ci) << 10) + (hh << 5) + ww;
        rBh[i] = inHi[gi]; rBl[i] = inLo[gi];
      } else { rBh[i] = 0; rBl[i] = 0; }
    }
  };
  auto storeB = [&](int bufw) {
#pragma unroll
    for (int i = 0; i < 16; ++i) {
      int e  = i * 256 + tid;
      int rr = e >> 5, kk = e & 31;
      sB[bufw][0][rr][kk] = rBh[i];
      sB[bufw][1][rr][kk] = rBl[i];
    }
  };

  // prologue: tile 0 into LDS[0]
  issueA(0, 0);
  loadB(0);
  storeA(0);
  storeB(0);
  wait_async0();
  __syncthreads();

  int buf = 0;
  for (int kb = 0; kb < KRED; kb += KSTEP) {
    const bool more = (kb + KSTEP) < KRED;
    if (more) {                         // next-tile loads overlap the WMMAs
      issueA(kb + KSTEP, buf ^ 1);      // async writes go straight to LDS
      loadB(kb + KSTEP);
    }

    // -- fragments: per 16-bit WMMA layouts, each lane's data is contiguous --
    // A lane data: K runs [8*hiHalf,+8) and [16+8*hiHalf,+8)  -> 2x b128
    // B lane data: K runs [8*hiHalf,+8) and [16+8*hiHalf,+8)  -> 2x b128
    FragU aF[2][2], bF[4][2];
#pragma unroll
    for (int t = 0; t < 2; ++t) {
      int row = mW + t * 16 + rlane;
#pragma unroll
      for (int hl = 0; hl < 2; ++hl) {
        const unsigned short* ap = &sA[buf][hl][row][hiHalf << 3];
        aF[t][hl].q[0] = *(const uint4*)ap;
        aF[t][hl].q[1] = *(const uint4*)(ap + 16);
      }
    }
#pragma unroll
    for (int t = 0; t < 4; ++t) {
      int col = nW + t * 16 + rlane;
#pragma unroll
      for (int hl = 0; hl < 2; ++hl) {
        const unsigned short* bp = &sB[buf][hl][col][hiHalf << 3];
        bF[t][hl].q[0] = *(const uint4*)bp;
        bF[t][hl].q[1] = *(const uint4*)(bp + 16);
      }
    }

    // -- 2x4 tiles x 3 split-WMMAs (hi*hi + hi*lo + lo*hi) ------------------
#pragma unroll
    for (int tm = 0; tm < 2; ++tm)
#pragma unroll
      for (int tn = 0; tn < 4; ++tn) {
        acc[tm][tn] = __builtin_amdgcn_wmma_f32_16x16x32_bf16(
            false, aF[tm][0].v, false, bF[tn][0].v, (short)0, acc[tm][tn], false, false);
        acc[tm][tn] = __builtin_amdgcn_wmma_f32_16x16x32_bf16(
            false, aF[tm][0].v, false, bF[tn][1].v, (short)0, acc[tm][tn], false, false);
        acc[tm][tn] = __builtin_amdgcn_wmma_f32_16x16x32_bf16(
            false, aF[tm][1].v, false, bF[tn][0].v, (short)0, acc[tm][tn], false, false);
      }

    // buf^1's last readers finished before the barrier that ended the
    // previous iteration, so one barrier per step is sufficient.
    if (more) { storeA(buf ^ 1); storeB(buf ^ 1); }
    wait_async0();
    __syncthreads();
    buf ^= 1;
  }

  // ---- epilogue: C/D layout lanes0-15 M=v, lanes16-31 M=v+8; N=lane%16 ----
#pragma unroll
  for (int tm = 0; tm < 2; ++tm) {
#pragma unroll
    for (int tn = 0; tn < 4; ++tn) {
      int p = nBase + nW + tn * 16 + rlane;
      int n = p >> 10, hw = p & 1023;
#pragma unroll
      for (int v = 0; v < 8; ++v) {
        int m   = mBase + mW + tm * 16 + v + (hiHalf ? 8 : 0);
        int idx = ((n * COUT + m) << 10) + hw;
        float val = acc[tm][tn][v];
        if (bn_relu_split) {
          float inv = rsqrtf(var[m] + 1e-5f);
          val = fmaxf((val - mu[m]) * (g[m] * inv) + b[m], 0.f);
          unsigned short hv = f2bf_rne(val);
          outHi[idx] = hv;
          outLo[idx] = f2bf_rne(val - bf2f(hv));
        } else {
          outF[idx] = val;
        }
      }
    }
  }
}

// ---------------------------------------------------------------------------
// per-descriptor L2 norm over channels (C=512)
// ---------------------------------------------------------------------------
__global__ __launch_bounds__(256)
void l2norm_c(const float* __restrict__ xe, float* __restrict__ xn)
{
  __shared__ float red[256];
  int p = blockIdx.x;
  int n = p >> 10, hw = p & 1023;
  const float* base = xe + (n * 512 << 10) + hw;
  int c0 = threadIdx.x, c1 = threadIdx.x + 256;
  float a  = base[c0 << 10];
  float bb = base[c1 << 10];
  red[threadIdx.x] = a * a + bb * bb;
  __syncthreads();
  for (int off = 128; off > 0; off >>= 1) {
    if (threadIdx.x < off) red[threadIdx.x] += red[threadIdx.x + off];
    __syncthreads();
  }
  float inv = 1.f / fmaxf(sqrtf(red[0]), L2EPS);
  float* ob = xn + (n * 512 << 10) + hw;
  ob[c0 << 10] = a * inv;
  ob[c1 << 10] = bb * inv;
}

// ---------------------------------------------------------------------------
// 1x1 assign conv + softmax over K=64 clusters (one block per descriptor)
// ---------------------------------------------------------------------------
__global__ __launch_bounds__(64)
void assign_softmax(const float* __restrict__ xn, const float* __restrict__ wA,
                    float* __restrict__ sa)
{
  __shared__ float sh[64];
  int p = blockIdx.x;
  int n = p >> 10, hw = p & 1023;
  int k = threadIdx.x;
  const float* xb = xn + (n * 512 << 10) + hw;
  const float* wb = wA + k * 512;
  float s = 0.f;
  for (int c = 0; c < 512; ++c) s += wb[c] * xb[c << 10];
  sh[k] = s; __syncthreads();
  for (int off = 32; off > 0; off >>= 1) {
    if (k < off) sh[k] = fmaxf(sh[k], sh[k + off]);
    __syncthreads();
  }
  float mx = sh[0]; __syncthreads();
  float e = expf(s - mx);
  sh[k] = e; __syncthreads();
  for (int off = 32; off > 0; off >>= 1) {
    if (k < off) sh[k] += sh[k + off];
    __syncthreads();
  }
  sa[(((n << 6) + k) << 10) | hw] = e / sh[0];
}

// ---------------------------------------------------------------------------
// ssum[n,k] = sum_l sa[n,k,l]
// ---------------------------------------------------------------------------
__global__ __launch_bounds__(256)
void sa_row_sum(const float* __restrict__ sa, float* __restrict__ ssum)
{
  __shared__ float red[256];
  int nk = blockIdx.x;
  const float* b = sa + (nk << 10);
  float s = 0.f;
  for (int i = threadIdx.x; i < 1024; i += 256) s += b[i];
  red[threadIdx.x] = s; __syncthreads();
  for (int off = 128; off > 0; off >>= 1) {
    if (threadIdx.x < off) red[threadIdx.x] += red[threadIdx.x + off];
    __syncthreads();
  }
  if (threadIdx.x == 0) ssum[nk] = red[0];
}

// ---------------------------------------------------------------------------
// vlad[n,k,c] = sum_l sa[n,k,l]*xn[n,c,l] - ssum[n,k]*cent[k,c]
// ---------------------------------------------------------------------------
__global__ __launch_bounds__(256)
void vlad_aggregate(const float* __restrict__ sa, const float* __restrict__ xn,
                    const float* __restrict__ ssum, const float* __restrict__ cent,
                    float* __restrict__ vpre)
{
  __shared__ float sS[1024];
  int n = blockIdx.x, k = blockIdx.y;
  const float* srow = sa + (((n << 6) + k) << 10);
  for (int i = threadIdx.x; i < 1024; i += 256) sS[i] = srow[i];
  __syncthreads();
  int c0 = threadIdx.x, c1 = threadIdx.x + 256;
  const float* x0 = xn + (n * 512 << 10) + (c0 << 10);
  const float* x1 = xn + (n * 512 << 10) + (c1 << 10);
  float a0 = 0.f, a1 = 0.f;
  for (int l = 0; l < 1024; ++l) {
    float s = sS[l];
    a0 += s * x0[l];
    a1 += s * x1[l];
  }
  float sm = ssum[(n << 6) + k];
  vpre[(n << 15) + (k << 9) + c0] = a0 - sm * cent[(k << 9) + c0];
  vpre[(n << 15) + (k << 9) + c1] = a1 - sm * cent[(k << 9) + c1];
}

// ---------------------------------------------------------------------------
// intra-normalization over c (512) per (n,k), in place
// ---------------------------------------------------------------------------
__global__ __launch_bounds__(256)
void intra_norm(float* __restrict__ vpre)
{
  __shared__ float red[256];
  int nk = blockIdx.x;
  float* b = vpre + (nk << 9);
  float a = b[threadIdx.x], c = b[threadIdx.x + 256];
  red[threadIdx.x] = a * a + c * c;
  __syncthreads();
  for (int off = 128; off > 0; off >>= 1) {
    if (threadIdx.x < off) red[threadIdx.x] += red[threadIdx.x + off];
    __syncthreads();
  }
  float inv = 1.f / fmaxf(sqrtf(red[0]), L2EPS);
  b[threadIdx.x]       = a * inv;
  b[threadIdx.x + 256] = c * inv;
}

// ---------------------------------------------------------------------------
// global L2 norm over 64*512 per batch element; writes final vlad output
// ---------------------------------------------------------------------------
__global__ __launch_bounds__(256)
void global_norm(const float* __restrict__ vpre, float* __restrict__ out)
{
  __shared__ float red[256];
  int n = blockIdx.x;
  const float* b = vpre + (n << 15);
  float s = 0.f;
  for (int i = threadIdx.x; i < 32768; i += 256) { float v = b[i]; s += v * v; }
  red[threadIdx.x] = s; __syncthreads();
  for (int off = 128; off > 0; off >>= 1) {
    if (threadIdx.x < off) red[threadIdx.x] += red[threadIdx.x + off];
    __syncthreads();
  }
  float inv = 1.f / fmaxf(sqrtf(red[0]), L2EPS);
  for (int i = threadIdx.x; i < 32768; i += 256)
    out[(n << 15) + i] = b[i] * inv;
}

// ---------------------------------------------------------------------------
extern "C" void kernel_launch(void* const* d_in, const int* in_sizes, int n_in,
                              void* d_out, int out_size, void* d_ws, size_t ws_size,
                              hipStream_t stream)
{
  const float* x    = (const float*)d_in[0];
  const float* w1   = (const float*)d_in[1];
  const float* g    = (const float*)d_in[2];
  const float* b    = (const float*)d_in[3];
  const float* mu   = (const float*)d_in[4];
  const float* var  = (const float*)d_in[5];
  const float* w2   = (const float*)d_in[6];
  const float* wA   = (const float*)d_in[7];
  const float* cent = (const float*)d_in[8];

  float* out      = (float*)d_out;
  float* vlad_out = out;                       // 16*64*512 = 524288 floats
  float* x_enc    = out + (16 * 64 * 512);     // 16*512*1024 floats

  // Workspace (peak ~73MB), phase-reused:
  //  R0 (32MB): xHi/xLo during conv1 -> w2Hi/w2Lo during conv2 -> xn after
  //  R1 (32MB): hHi/hLo during convs -> sa/ssum/vpre after
  //  R2 (9MB):  w1Hi/w1Lo
  char* wsb = (char*)d_ws;
  unsigned short* xHi  = (unsigned short*)(wsb);
  unsigned short* xLo  = (unsigned short*)(wsb + 16777216);
  unsigned short* w2Hi = (unsigned short*)(wsb);                 // after conv1
  unsigned short* w2Lo = (unsigned short*)(wsb + 4718592);
  float*          xn   = (float*)(wsb);                          // after conv2

  unsigned short* hHi  = (unsigned short*)(wsb + 33554432);
  unsigned short* hLo  = (unsigned short*)(wsb + 33554432 + 16777216);
  float*          sa   = (float*)(wsb + 33554432);               // after conv2
  float*          ssum = sa + 1048576;
  float*          vpre = ssum + 1024;

  unsigned short* w1Hi = (unsigned short*)(wsb + 67108864);
  unsigned short* w1Lo = (unsigned short*)(wsb + 67108864 + 4718592);

  const int WELEMS = 512 * 512 * 9;   // 2359296
  const int XELEMS = 16 * 512 * 1024; // 8388608

  split_bf16<<<(WELEMS + 255) / 256, 256, 0, stream>>>(w1, w1Hi, w1Lo, WELEMS);
  split_bf16<<<(XELEMS + 255) / 256, 256, 0, stream>>>(x, xHi, xLo, XELEMS);

  // conv1 + BN + ReLU, epilogue emits h as bf16 hi/lo split
  conv3x3_wmma<<<dim3(512 / MT, NPIX / NT), 256, 0, stream>>>(
      xHi, xLo, w1Hi, w1Lo, nullptr, hHi, hLo, g, b, mu, var, 1);

  split_bf16<<<(WELEMS + 255) / 256, 256, 0, stream>>>(w2, w2Hi, w2Lo, WELEMS);

  // conv2 -> x_enc (second tuple output, f32)
  conv3x3_wmma<<<dim3(512 / MT, NPIX / NT), 256, 0, stream>>>(
      hHi, hLo, w2Hi, w2Lo, x_enc, nullptr, nullptr, g, b, mu, var, 0);

  l2norm_c<<<NPIX, 256, 0, stream>>>(x_enc, xn);
  assign_softmax<<<NPIX, 64, 0, stream>>>(xn, wA, sa);
  sa_row_sum<<<1024, 256, 0, stream>>>(sa, ssum);
  vlad_aggregate<<<dim3(16, 64), 256, 0, stream>>>(sa, xn, ssum, cent, vpre);
  intra_norm<<<1024, 256, 0, stream>>>(vpre);
  global_norm<<<16, 256, 0, stream>>>(vpre, vlad_out);
}
